// ST_BLOCK_8_14035953123745
// MI455X (gfx1250) — matrix operand order repaired and checked by reading, not compiled, not run on hardware
//
#include <hip/hip_runtime.h>

// ---------------------------------------------------------------------------
// ST-Block for MI455X (gfx1250, wave32, WMMA).
// Dominant cost: 8 GEMMs of (1024x1024)x(1024x31744) in bf16 -> f32 via
// v_wmma_f32_16x16x32_bf16.  All node-tensors kept transposed [m][node]
// so WMMA B-fragments are two k-contiguous 16B per-lane runs.
// GEMM: 128(node) x 256(col) workgroup tile staged in LDS, double-buffered
// with GLOBAL_LOAD_ASYNC_TO_LDS_B128 (inline asm; ASYNCcnt-tracked),
// 64x64 per wave = 16 f32 accumulators.
// ---------------------------------------------------------------------------

typedef __bf16 bf16;
typedef __attribute__((ext_vector_type(16))) __bf16 v16bf;
typedef __attribute__((ext_vector_type(8)))  __bf16 v8bf;
typedef __attribute__((ext_vector_type(8)))  float  v8f;

#define NNODE 1024
#define BB    16
#define CC    32
#define LL    62
#define MTOT  (BB * LL * CC)   // 31744

#define USE_ASYNC_ASM 1

#if USE_ASYNC_ASM
// Async DMA: global -> LDS, 16B per lane, tracked by ASYNCcnt.
//   VDST  = per-lane LDS byte offset (generic LDS ptr truncated to 32 bits;
//           flat LDS addresses use the low 32 bits)
//   VADDR = 64-bit global address (VGPR pair)
__device__ __forceinline__ void async_load_b128(const bf16* g, const bf16* l) {
  asm volatile("global_load_async_to_lds_b128 %0, %1, off"
               :
               : "v"((unsigned)(size_t)l), "v"(g)
               : "memory");
}
__device__ __forceinline__ void wait_asynccnt_6() {
  asm volatile("s_wait_asynccnt 6" ::: "memory");
}
__device__ __forceinline__ void wait_asynccnt_0() {
  asm volatile("s_wait_asynccnt 0" ::: "memory");
}
#endif

__device__ __forceinline__ v8f wmma_bf16(v16bf a, v16bf b, v8f c) {
  return __builtin_amdgcn_wmma_f32_16x16x32_bf16(
      /*neg_a=*/false, a, /*neg_b=*/false, b,
      /*c_mod=*/(short)0, c, /*reuse_a=*/false, /*reuse_b=*/false);
}

// A-fragment (16x32, M x K).  Lane layout per ISA 7.12.2:
//   row = (lane&15); K = (lane>=16?8:0) + (e>=8?16:0) + (e&7)
__device__ __forceinline__ v16bf load_frag_a(const bf16* p, int ldk, int row0,
                                             int k0, int lane) {
  int r  = row0 + (lane & 15);
  int kb = k0 + ((lane >> 4) & 1) * 8;
  const bf16* q = p + (size_t)r * ldk + kb;
  v8bf lo = *(const v8bf*)(q);       // K = kb .. kb+7
  v8bf hi = *(const v8bf*)(q + 16);  // K = kb+16 .. kb+23
  v16bf out;
#pragma unroll
  for (int i = 0; i < 8; ++i) { out[i] = lo[i]; out[8 + i] = hi[i]; }
  return out;
}

// B-fragment (32x16, K x N).  lane col = (lane&15); K = (lane>=16?8:0) + (e>=8?16:0) + (e&7).
// Memory layout is [col][k] (k contiguous) -> two 16B per-lane runs.
__device__ __forceinline__ v16bf load_frag_b(const bf16* p, int ldk, int col0,
                                             int k0, int lane) {
  int c  = col0 + (lane & 15);
  int kb = k0 + ((lane >> 4) & 1) * 8;
  const bf16* q = p + (size_t)c * ldk + kb;
  v8bf lo = *(const v8bf*)(q);
  v8bf hi = *(const v8bf*)(q + 16);
  v16bf out;
#pragma unroll
  for (int i = 0; i < 8; ++i) { out[i] = lo[i]; out[8 + i] = hi[i]; }
  return out;
}

// ---------------------------------------------------------------------------
// Kernel 0: cast supports + weights to bf16, permute Wg, fold BN.
// ---------------------------------------------------------------------------
__global__ void prep_consts(const float* __restrict__ S,
                            const float* __restrict__ A0,
                            const float* __restrict__ A1,
                            const float* __restrict__ A2,
                            const float* __restrict__ Wg,
                            const float* __restrict__ Wm,
                            const float* __restrict__ W1,
                            const float* __restrict__ gamma,
                            const float* __restrict__ beta,
                            const float* __restrict__ mean,
                            const float* __restrict__ var,
                            bf16* __restrict__ Sb, bf16* __restrict__ A0b,
                            bf16* __restrict__ A1b, bf16* __restrict__ A2b,
                            bf16* __restrict__ WgT, bf16* __restrict__ WmT,
                            bf16* __restrict__ W1T,
                            float* __restrict__ bnscale,
                            float* __restrict__ bnshift) {
  int i = blockIdx.x * blockDim.x + threadIdx.x;
  const int NN2 = NNODE * NNODE;
  if (i < NN2) {
    Sb[i]  = (bf16)S[i];
    A0b[i] = (bf16)A0[i];
    A1b[i] = (bf16)A1[i];
    A2b[i] = (bf16)A2[i];
  }
  if (i < 64 * 96) {  // Wg: interleaved channel c*3+src -> block src*32+c
    int o = i / 96, k = i % 96;
    int src = k / 32, c = k % 32;
    WgT[o * 96 + k] = (bf16)Wg[o * 96 + c * 3 + src];
  }
  if (i < 64 * 224) WmT[i] = (bf16)Wm[i];  // already block order [o][k]
  if (i < 64 * 32)  W1T[i] = (bf16)W1[i];  // [o][c]
  if (i < 64) {
    float inv = gamma[i] * rsqrtf(var[i] + 1e-5f);
    bnscale[i] = inv;
    bnshift[i] = beta[i] - mean[i] * inv;
  }
}

// ---------------------------------------------------------------------------
// Kernel 1: time conv (1xKT, VALID) + bias -> XT[m][node] bf16.
// ---------------------------------------------------------------------------
__global__ __launch_bounds__(256) void prep_xt(const float* __restrict__ x,
                                               const float* __restrict__ Wt,
                                               const float* __restrict__ bt,
                                               bf16* __restrict__ XT) {
  int bl = blockIdx.x;           // (b*LL + l)
  int b = bl / LL, l = bl % LL;
  __shared__ float w[32 * 32 * 3];
  __shared__ float bsh[32];
  for (int i = threadIdx.x; i < 32 * 32 * 3; i += 256) w[i] = Wt[i];
  if (threadIdx.x < 32) bsh[threadIdx.x] = bt[threadIdx.x];
  __syncthreads();
  for (int n = (int)threadIdx.x; n < NNODE; n += 256) {
    float xin[32][3];
#pragma unroll
    for (int ci = 0; ci < 32; ++ci) {
      const float* px = x + (((size_t)b * 32 + ci) * NNODE + n) * 64 + l;
#pragma unroll
      for (int kt = 0; kt < 3; ++kt) xin[ci][kt] = px[kt];
    }
#pragma unroll 4
    for (int co = 0; co < 32; ++co) {
      float s = bsh[co];
#pragma unroll
      for (int ci = 0; ci < 32; ++ci)
#pragma unroll
        for (int kt = 0; kt < 3; ++kt)
          s = fmaf(xin[ci][kt], w[(co * 32 + ci) * 3 + kt], s);
      XT[((size_t)(bl * 32 + co)) * NNODE + n] = (bf16)s;
    }
  }
}

// ---------------------------------------------------------------------------
// Kernel 2: graph GEMM  Y[m][:] = alpha*(G @ X[m][:]) + beta*Z[m][:]
//   G : [1024 x 1024] bf16 row-major        (A operand)
//   X : [m][node] bf16 (transposed layout)  (B operand)
// Workgroup: 256 threads / 8 waves; WG tile 128 nodes x 256 cols.
// K-step tiles (A:128x32 = 8KB, B:256x32 = 16KB) double-buffered in LDS via
// global_load_async_to_lds_b128 (6 x 16B chunks per thread per slice).
// Each wave: 64 nodes x 64 cols = 16 f32 accumulators, 16 WMMA / K-step.
// ---------------------------------------------------------------------------
__global__ __launch_bounds__(256) void graph_gemm(
    const bf16* __restrict__ G, const bf16* __restrict__ X,
    bf16* __restrict__ Y, const bf16* __restrict__ Z,
    float alpha, float beta) {
  __shared__ bf16 Abuf[2][128 * 32];   // [node_local][k]
  __shared__ bf16 Bbuf[2][256 * 32];   // [col_local][k]
  const int tid  = threadIdx.x;
  const int lane = tid & 31;
  const int wave = tid >> 5;
  const int nodeWG = blockIdx.y * 128;
  const int colWG  = blockIdx.x * 256;
  const int nodeBase = (wave >> 2) * 64;   // within WG tile
  const int colBase  = (wave & 3) * 64;

  v8f acc[4][4] = {};

#if USE_ASYNC_ASM
  // Each thread owns 6 x 16B chunks per K-slice: 2 of A (512 chunks total),
  // 4 of B (1024 chunks).  ASYNCcnt per wave per slice = 6.
  auto issue_tile = [&](int buf, int k0) {
    const bf16* gA = G + (size_t)nodeWG * NNODE + k0;
    bf16* lA = &Abuf[buf][0];
#pragma unroll
    for (int j = 0; j < 2; ++j) {
      int idx = tid + j * 256;
      int r = idx >> 2, c = (idx & 3) * 8;            // 8 bf16 = 16B
      async_load_b128(gA + (size_t)r * NNODE + c, lA + r * 32 + c);
    }
    const bf16* gB = X + (size_t)colWG * NNODE + k0;
    bf16* lB = &Bbuf[buf][0];
#pragma unroll
    for (int j = 0; j < 4; ++j) {
      int idx = tid + j * 256;
      int col = idx >> 2, c = (idx & 3) * 8;
      async_load_b128(gB + (size_t)col * NNODE + c, lB + col * 32 + c);
    }
  };

  issue_tile(0, 0);
#pragma unroll 1
  for (int k0 = 0; k0 < NNODE; k0 += 32) {
    const int cur = (k0 >> 5) & 1;
    const bool more = (k0 + 32) < NNODE;
    if (more) {
      issue_tile(cur ^ 1, k0 + 32);
      wait_asynccnt_6();                 // current slice complete (in-order)
    } else {
      wait_asynccnt_0();
    }
    __syncthreads();                     // all waves' slice visible
    const bf16* At = &Abuf[cur][0];
    const bf16* Bt = &Bbuf[cur][0];
    v16bf af[4], bf[4];
#pragma unroll
    for (int i = 0; i < 4; ++i) af[i] = load_frag_a(At, 32, nodeBase + i * 16, 0, lane);
#pragma unroll
    for (int j = 0; j < 4; ++j) bf[j] = load_frag_b(Bt, 32, colBase + j * 16, 0, lane);
#pragma unroll
    for (int i = 0; i < 4; ++i)
#pragma unroll
      for (int j = 0; j < 4; ++j) acc[i][j] = wmma_bf16(af[i], bf[j], acc[i][j]);
    __syncthreads();                     // done reading before overwrite
  }
#else
  // Synchronous fallback: load -> LDS -> barrier -> consume (single buffer).
#pragma unroll 1
  for (int k0 = 0; k0 < NNODE; k0 += 32) {
    const bf16* gA = G + (size_t)nodeWG * NNODE + k0;
    const bf16* gB = X + (size_t)colWG * NNODE + k0;
#pragma unroll
    for (int j = 0; j < 2; ++j) {
      int idx = tid + j * 256;
      int r = idx >> 2, c = (idx & 3) * 8;
      *(v8bf*)&Abuf[0][r * 32 + c] = *(const v8bf*)(gA + (size_t)r * NNODE + c);
    }
#pragma unroll
    for (int j = 0; j < 4; ++j) {
      int idx = tid + j * 256;
      int col = idx >> 2, c = (idx & 3) * 8;
      *(v8bf*)&Bbuf[0][col * 32 + c] = *(const v8bf*)(gB + (size_t)col * NNODE + c);
    }
    __syncthreads();
    const bf16* At = &Abuf[0][0];
    const bf16* Bt = &Bbuf[0][0];
    v16bf af[4], bf[4];
#pragma unroll
    for (int i = 0; i < 4; ++i) af[i] = load_frag_a(At, 32, nodeBase + i * 16, 0, lane);
#pragma unroll
    for (int j = 0; j < 4; ++j) bf[j] = load_frag_b(Bt, 32, colBase + j * 16, 0, lane);
#pragma unroll
    for (int i = 0; i < 4; ++i)
#pragma unroll
      for (int j = 0; j < 4; ++j) acc[i][j] = wmma_bf16(af[i], bf[j], acc[i][j]);
    __syncthreads();
  }
#endif

  // D layout: lane col = lane&15, nodes (r + hi*8) -> 8 consecutive nodes/lane
  const int hi = (lane >> 4) & 1;
#pragma unroll
  for (int i = 0; i < 4; ++i) {
#pragma unroll
    for (int j = 0; j < 4; ++j) {
      int m = colWG + colBase + j * 16 + (lane & 15);
      int n = nodeWG + nodeBase + i * 16 + hi * 8;
      bf16* q = Y + (size_t)m * NNODE + n;
      v8bf o;
      if (Z != nullptr) {
        v8bf z = *(const v8bf*)(Z + (size_t)m * NNODE + n);
#pragma unroll
        for (int e = 0; e < 8; ++e)
          o[e] = (bf16)(alpha * acc[i][j][e] + beta * (float)z[e]);
      } else {
#pragma unroll
        for (int e = 0; e < 8; ++e) o[e] = (bf16)(alpha * acc[i][j][e]);
      }
      *(v8bf*)q = o;
    }
  }
}

// ---------------------------------------------------------------------------
// Kernel 3: fused epilogue (WMMA 1x1 convs + GLU + residual + BN).
// Block tile: fixed b, 8 nodes x 8 l -> 64 positions (pos = node_local*8+dl).
//   waves 0-3 : Wg GEMM (K=96: xt|y1c|y2c) + W1 residual (o 0..31)
//   waves 4-7 : Wm GEMM (K=224: xt|x1_0|x2_0|x1_1|x2_1|x1_2|x2_2) + W1 (o 32..63)
// ---------------------------------------------------------------------------
__global__ __launch_bounds__(256) void epilogue_kernel(
    const bf16* __restrict__ XT,  const bf16* __restrict__ Y1C,
    const bf16* __restrict__ Y2C, const bf16* __restrict__ X10,
    const bf16* __restrict__ X20, const bf16* __restrict__ X11,
    const bf16* __restrict__ X21, const bf16* __restrict__ X12,
    const bf16* __restrict__ X22, const float* __restrict__ x,
    const bf16* __restrict__ WgT, const bf16* __restrict__ WmT,
    const bf16* __restrict__ W1T, const float* __restrict__ bg,
    const float* __restrict__ bm, const float* __restrict__ b1,
    const float* __restrict__ bnscale, const float* __restrict__ bnshift,
    float* __restrict__ out) {
  __shared__ bf16 In[64][320];  // 40 KB: [pos(node*8+dl)][chan]
  const int l0    = blockIdx.x * 8;
  const int node0 = blockIdx.y * 8;
  const int b     = blockIdx.z;

  const bf16* tensors[9] = {XT, Y1C, Y2C, X10, X20, X11, X21, X12, X22};
  for (int row = (int)threadIdx.x; row < 9 * 256; row += 256) {
    int t = row >> 8, c = (row >> 3) & 31, dl = row & 7;
    int dle = l0 + dl; if (dle > LL - 1) dle = LL - 1;     // clamp tail
    size_t m = (size_t)(b * LL + dle) * 32 + c;
    v8bf v = *(const v8bf*)(tensors[t] + m * NNODE + node0);
#pragma unroll
    for (int nl = 0; nl < 8; ++nl) In[nl * 8 + dl][t * 32 + c] = v[nl];
  }
  for (int row = (int)threadIdx.x; row < 256; row += 256) {
    int c = row >> 3, dl = row & 7;
    int tt = l0 + dl + 2; if (tt > 63) tt = 63;            // t = l+2 (last-L)
#pragma unroll
    for (int nl = 0; nl < 8; ++nl) {
      float xv = x[(((size_t)b * 32 + c) * NNODE + node0 + nl) * 64 + tt];
      In[nl * 8 + dl][288 + c] = (bf16)xv;
    }
  }
  __syncthreads();

  const int lane = threadIdx.x & 31;
  const int wave = threadIdx.x >> 5;
  const int st   = wave & 3;        // position subtile (16 pos = 2 nodes x 8 l)
  const bool grpB = wave >= 4;
  const int rowBase = st * 16;
  const bf16* lds = &In[0][0];

  v8f g[4] = {};
  v8f r[2] = {};

  if (!grpB) {
#pragma unroll
    for (int kb = 0; kb < 3; ++kb) {
      v16bf a = load_frag_a(lds, 320, rowBase, kb * 32, lane);
#pragma unroll
      for (int nt = 0; nt < 4; ++nt) {
        v16bf bf = load_frag_b(WgT, 96, nt * 16, kb * 32, lane);
        g[nt] = wmma_bf16(a, bf, g[nt]);
      }
    }
    {
      v16bf a = load_frag_a(lds, 320, rowBase, 288, lane);
#pragma unroll
      for (int nt = 0; nt < 2; ++nt) {
        v16bf bf = load_frag_b(W1T, 32, nt * 16, 0, lane);
        r[nt] = wmma_bf16(a, bf, r[nt]);
      }
    }
  } else {
    const int kblk[7] = {0, 96, 128, 160, 192, 224, 256};
#pragma unroll
    for (int kb = 0; kb < 7; ++kb) {
      v16bf a = load_frag_a(lds, 320, rowBase, kblk[kb], lane);
#pragma unroll
      for (int nt = 0; nt < 4; ++nt) {
        v16bf bf = load_frag_b(WmT, 224, nt * 16, kb * 32, lane);
        g[nt] = wmma_bf16(a, bf, g[nt]);
      }
    }
    {
      v16bf a = load_frag_a(lds, 320, rowBase, 288, lane);
#pragma unroll
      for (int nt = 0; nt < 2; ++nt) {
        v16bf bf = load_frag_b(W1T, 32, 32 + nt * 16, 0, lane);
        r[nt] = wmma_bf16(a, bf, r[nt]);
      }
    }
  }

  const float* gb = grpB ? bm : bg;
#pragma unroll
  for (int nt = 0; nt < 4; ++nt) {
    float bb = gb[nt * 16 + (lane & 15)];
#pragma unroll
    for (int i = 0; i < 8; ++i) g[nt][i] += bb;
  }
#pragma unroll
  for (int nt = 0; nt < 2; ++nt) {
    float bb = b1[(grpB ? 32 : 0) + nt * 16 + (lane & 15)];
#pragma unroll
    for (int i = 0; i < 8; ++i) r[nt][i] += bb;
  }

  const int hi = (lane >> 4) & 1;
  const int node = node0 + st * 2 + hi;
#pragma unroll
  for (int nt = 0; nt < 2; ++nt) {
    int oc = (grpB ? 32 : 0) + nt * 16 + (lane & 15);
    float sc = bnscale[oc], sh = bnshift[oc];
    float* po = out + (((size_t)b * 64 + oc) * NNODE + node) * LL + l0;
#pragma unroll
    for (int i = 0; i < 8; ++i) {
      if (l0 + i < LL) {
        float gv = fmaxf(g[nt][i], 0.f);
        float sv = 1.f / (1.f + __expf(-g[nt + 2][i]));
        po[i] = (gv * sv + r[nt][i]) * sc + sh;
      }
    }
  }
}

// ---------------------------------------------------------------------------
// Host launcher.  Workspace budget: ~594 MB
// ---------------------------------------------------------------------------
extern "C" void kernel_launch(void* const* d_in, const int* in_sizes, int n_in,
                              void* d_out, int out_size, void* d_ws,
                              size_t ws_size, hipStream_t stream) {
  (void)in_sizes; (void)n_in; (void)out_size; (void)ws_size;
  const float* x   = (const float*)d_in[0];
  const float* S   = (const float*)d_in[1];
  const float* A0  = (const float*)d_in[2];
  const float* A1  = (const float*)d_in[3];
  const float* A2  = (const float*)d_in[4];
  const float* W1  = (const float*)d_in[5];
  const float* b1  = (const float*)d_in[6];
  const float* Wt  = (const float*)d_in[7];
  const float* bt  = (const float*)d_in[8];
  const float* Wg  = (const float*)d_in[9];
  const float* bg  = (const float*)d_in[10];
  const float* Wm  = (const float*)d_in[11];
  const float* bm  = (const float*)d_in[12];
  const float* gam = (const float*)d_in[13];
  const float* bet = (const float*)d_in[14];
  const float* rmu = (const float*)d_in[15];
  const float* rvr = (const float*)d_in[16];
  float* out = (float*)d_out;

  uint8_t* w = (uint8_t*)d_ws;
  size_t off = 0;
  auto alloc = [&](size_t bytes) {
    size_t o = off;
    off = (off + bytes + 255) & ~(size_t)255;
    return o;
  };
  const size_t NN2B = (size_t)NNODE * NNODE * sizeof(bf16);
  const size_t TSB  = (size_t)MTOT * NNODE * sizeof(bf16);

  bf16* Sb  = (bf16*)(w + alloc(NN2B));
  bf16* A0b = (bf16*)(w + alloc(NN2B));
  bf16* A1b = (bf16*)(w + alloc(NN2B));
  bf16* A2b = (bf16*)(w + alloc(NN2B));
  bf16* WgT = (bf16*)(w + alloc(64 * 96 * sizeof(bf16)));
  bf16* WmT = (bf16*)(w + alloc(64 * 224 * sizeof(bf16)));
  bf16* W1T = (bf16*)(w + alloc(64 * 32 * sizeof(bf16)));
  float* bns = (float*)(w + alloc(64 * sizeof(float)));
  float* bnh = (float*)(w + alloc(64 * sizeof(float)));
  bf16* XT  = (bf16*)(w + alloc(TSB));
  bf16* Y1C = (bf16*)(w + alloc(TSB));
  bf16* Y2C = (bf16*)(w + alloc(TSB));
  bf16* X10 = (bf16*)(w + alloc(TSB));
  bf16* X20 = (bf16*)(w + alloc(TSB));
  bf16* X11 = (bf16*)(w + alloc(TSB));
  bf16* X21 = (bf16*)(w + alloc(TSB));
  bf16* X12 = (bf16*)(w + alloc(TSB));
  bf16* X22 = (bf16*)(w + alloc(TSB));

  prep_consts<<<(NNODE * NNODE + 255) / 256, 256, 0, stream>>>(
      S, A0, A1, A2, Wg, Wm, W1, gam, bet, rmu, rvr,
      Sb, A0b, A1b, A2b, WgT, WmT, W1T, bns, bnh);

  prep_xt<<<BB * LL, 256, 0, stream>>>(x, Wt, bt, XT);

  // hop-1 GEMMs (all read XT: L2 reuse across launches)
  dim3 ggrid(MTOT / 256, NNODE / 128);
  graph_gemm<<<ggrid, 256, 0, stream>>>(Sb,  XT, Y1C, nullptr, 1.f, 0.f);
  graph_gemm<<<ggrid, 256, 0, stream>>>(A0b, XT, X10, nullptr, 1.f, 0.f);
  graph_gemm<<<ggrid, 256, 0, stream>>>(A1b, XT, X11, nullptr, 1.f, 0.f);
  graph_gemm<<<ggrid, 256, 0, stream>>>(A2b, XT, X12, nullptr, 1.f, 0.f);

  // hop-2 GEMMs (y2c = 2*S@y1c - xt folds the Chebyshev recurrence)
  graph_gemm<<<ggrid, 256, 0, stream>>>(Sb,  Y1C, Y2C, XT, 2.f, -1.f);
  graph_gemm<<<ggrid, 256, 0, stream>>>(A0b, X10, X20, nullptr, 1.f, 0.f);
  graph_gemm<<<ggrid, 256, 0, stream>>>(A1b, X11, X21, nullptr, 1.f, 0.f);
  graph_gemm<<<ggrid, 256, 0, stream>>>(A2b, X12, X22, nullptr, 1.f, 0.f);

  dim3 egrid(8 /*l chunks*/, NNODE / 8, BB);
  epilogue_kernel<<<egrid, 256, 0, stream>>>(
      XT, Y1C, Y2C, X10, X20, X11, X21, X12, X22, x,
      WgT, WmT, W1T, bg, bm, b1, bns, bnh, out);
}
